// regularization_GlobalPointer_35304631173233
// MI455X (gfx1250) — compile-verified, matrix operand order checked by reading, not run on hardware
//
#include <hip/hip_runtime.h>
#include <cstdint>
#include <cstddef>

// Problem constants (match reference)
#define HIDDEN 768
#define ENT    9
#define DHEAD  64
#define SEQ    512
#define NB     16
#define NCOL   (ENT * 2 * DHEAD)   /* 1152 */
#define BIGC   1000000000000.0f

typedef __attribute__((ext_vector_type(16))) _Float16 v16h;
typedef __attribute__((ext_vector_type(8)))  float    v8f;

#if defined(__has_builtin)
#if __has_builtin(__builtin_amdgcn_tensor_load_to_lds) && \
    __has_builtin(__builtin_amdgcn_s_wait_tensorcnt)
#define HAVE_TDM 1
#endif
#endif

union FragU { v16h v; uint4 q[2]; };

// Load one 16x32 (or 32x16) f16 operand fragment from an LDS row.
// CDNA5 16-bit A layout: lanes 0-15 hold K[kOff..kOff+8) and K[kOff+16..kOff+24)
// (kOff = 0 for lanes 0-15, 8 for lanes 16-31), packed 2 halves per VGPR.
__device__ __forceinline__ v16h load_frag(const _Float16* row, int kOff) {
  FragU f;
  f.q[0] = *(const uint4*)(row + kOff);
  f.q[1] = *(const uint4*)(row + kOff + 16);
  return f.v;
}

__device__ __forceinline__ v8f wmma16(v16h a, v16h b, v8f c) {
  // 8 args: (neg_a, A, neg_b, B, c_mod, C, reuse_a, reuse_b)
  return __builtin_amdgcn_wmma_f32_16x16x32_f16(false, a, false, b, (short)0, c,
                                                false, false);
}

#ifdef HAVE_TDM
typedef __attribute__((ext_vector_type(4))) unsigned int u32x4;
typedef __attribute__((ext_vector_type(4))) int          i32x4;
typedef __attribute__((ext_vector_type(8))) int          i32x8;

// TDM: copy a 128-row x 128-byte 2D tile (contiguous in global) into LDS,
// inserting 16B of padding after every 128B row (-> 144B LDS row stride).
// D# packing per CDNA5 ISA ch.8 (group0 128b, group1 256b; groups 2/3 unused
// for 2D). data_size = 8B units: row = 16 units = 32 DWORDs; pad_interval
// code 4 = 32 DWORDs, pad_amount code 3 = 4 DWORDs (16B).
__device__ __forceinline__ void tdm_load_tile_pad(const void* gsrc, void* ldsDst) {
  const unsigned long long ga = (unsigned long long)(uintptr_t)gsrc;
  const unsigned lds = (unsigned)(uintptr_t)ldsDst;  // low 32b of generic = LDS byte addr
  u32x4 g0;
  g0.x = 1u;                                              // count=1 (valid user D#)
  g0.y = lds;                                             // lds_addr
  g0.z = (unsigned)ga;                                    // global_addr[31:0]
  g0.w = ((unsigned)(ga >> 32) & 0x01ffffffu) | (2u << 30); // addr[56:32] | type=2
  i32x8 g1;
  g1.s0 = (3 << 16) | (1 << 20) | (4 << 22) | (3 << 25);  // data_size=8B, pad on
  g1.s1 = (16 << 16);                                     // tensor_dim0 = 16 units
  g1.s2 = (128 << 16);                                    // tensor_dim1 = 128 rows
  g1.s3 = (16 << 16);                                     // tile_dim0 = 16 units
  g1.s4 = 128;                                            // tile_dim1 = 128 (tile_dim2=0)
  g1.s5 = 16;                                             // tensor_dim0_stride = 16 units
  g1.s6 = 0;
  g1.s7 = 0;
  i32x4 z4 = {0, 0, 0, 0};
#if __clang_major__ >= 23
  i32x8 z8 = {0, 0, 0, 0, 0, 0, 0, 0};
  __builtin_amdgcn_tensor_load_to_lds(g0, g1, z4, z4, z8, 0);
#else
  __builtin_amdgcn_tensor_load_to_lds(g0, g1, z4, z4, 0);
#endif
}
#endif  // HAVE_TDM

// ---------------------------------------------------------------------------
// Kernel 1: out = X @ W + b, split q/k, RoPE, store f16 qh/kh [b,e,s,d]
// Grid: (8192/128, 1152/128) = (64, 9). Block: 256 (8 waves).
// ---------------------------------------------------------------------------
__global__ __launch_bounds__(256) void gemm1_rope_kernel(
    const float* __restrict__ X,     // (8192, 768)
    const float* __restrict__ W,     // (768, 1152)
    const float* __restrict__ bias,  // (1152)
    _Float16* __restrict__ qh,       // (16*9*512, 64)
    _Float16* __restrict__ kh)
{
  __shared__ __align__(16) _Float16 As[128 * 40];  // [m][k] stride 40 halves (80B, 16B-aligned)
  __shared__ __align__(16) _Float16 Bs[128 * 40];  // [n][k] (transposed store)

  const int tid   = threadIdx.x;
  const int lane  = tid & 31;
  const int waveM = (tid >> 5) & 3;   // 32 rows each
  const int waveN = tid >> 7;         // 64 cols each
  const int m0 = blockIdx.x * 128;
  const int n0 = blockIdx.y * 128;

  v8f acc[2][4];
  {
    v8f z = {};
    #pragma unroll
    for (int i = 0; i < 2; ++i)
      #pragma unroll
      for (int j = 0; j < 4; ++j) acc[i][j] = z;
  }

  const int rowL = lane & 15;
  const int kSel = (lane >> 4) * 8;

  for (int k0 = 0; k0 < HIDDEN; k0 += 32) {
    __syncthreads();
    // Stage A: 128x32 f32 -> f16 (coalesced along k)
    #pragma unroll
    for (int i = 0; i < 16; ++i) {
      int idx = tid + i * 256;
      int r   = idx >> 5;
      int kk  = idx & 31;
      As[r * 40 + kk] = (_Float16)X[(size_t)(m0 + r) * HIDDEN + k0 + kk];
    }
    // Stage B transposed: 32x128 f32 -> f16 LDS[n][k] (coalesced along n)
    #pragma unroll
    for (int i = 0; i < 16; ++i) {
      int idx = tid + i * 256;
      int kk  = idx >> 7;
      int nn  = idx & 127;
      Bs[nn * 40 + kk] = (_Float16)W[(size_t)(k0 + kk) * NCOL + n0 + nn];
    }
    if (k0 + 32 < HIDDEN)  // hint next A tile into cache (global_prefetch_b8)
      __builtin_prefetch(&X[(size_t)(m0 + (tid >> 1)) * HIDDEN + k0 + 32], 0, 1);
    __syncthreads();

    v16h a[2], b[4];
    #pragma unroll
    for (int mt = 0; mt < 2; ++mt)
      a[mt] = load_frag(&As[(waveM * 32 + mt * 16 + rowL) * 40], kSel);
    #pragma unroll
    for (int nt = 0; nt < 4; ++nt)
      b[nt] = load_frag(&Bs[(waveN * 64 + nt * 16 + rowL) * 40], kSel);
    #pragma unroll
    for (int mt = 0; mt < 2; ++mt)
      #pragma unroll
      for (int nt = 0; nt < 4; ++nt)
        acc[mt][nt] = wmma16(a[mt], b[nt], acc[mt][nt]);
  }

  // Epilogue: bias + RoPE (interleaved pairs: partner value sits in lane^1)
  #pragma unroll
  for (int mt = 0; mt < 2; ++mt) {
    #pragma unroll
    for (int nt = 0; nt < 4; ++nt) {
      const int ng = n0 + waveN * 64 + nt * 16 + rowL;   // 0..1151
      const int e  = ng >> 7;
      const int c  = ng & 127;
      const int d  = c & 63;
      _Float16* dst = (c >= 64) ? kh : qh;
      const float invf = __powf(10000.0f, -(float)(d & ~1) * (1.0f / 64.0f));
      const float bn = bias[ng];
      #pragma unroll
      for (int r = 0; r < 8; ++r) {
        const int mg = m0 + waveM * 32 + mt * 16 + ((lane >> 4) << 3) + r;
        const int s  = mg & (SEQ - 1);
        const int bb = mg >> 9;
        float v = acc[mt][nt][r] + bn;
        float p = __shfl_xor(v, 1, 32);          // partner dim (d^1)
        float sn, cs;
        __sincosf((float)s * invf, &sn, &cs);
        float o = (d & 1) ? (v * cs + p * sn) : (v * cs - p * sn);
        size_t idx = (((size_t)(bb * ENT + e) * SEQ) + s) * DHEAD + d;
        dst[idx] = (_Float16)o;
      }
    }
  }
}

// ---------------------------------------------------------------------------
// Kernel 2: logits[b,e,m,n] = (q . k) masked / 8, fp32 out.
// Grid: (16 tiles of 128x128, 144 batches). Block: 256.
// Q/K tiles staged into LDS by the Tensor Data Mover (wave 0 issues two 2D
// tile descriptors with LDS padding; s_wait_tensorcnt 0 before the barrier).
// ---------------------------------------------------------------------------
__global__ __launch_bounds__(256) void qk_logits_kernel(
    const _Float16* __restrict__ qh,
    const _Float16* __restrict__ kh,
    const int* __restrict__ amask,    // (16,512)
    float* __restrict__ logits)       // (144,512,512)
{
  __shared__ __align__(16) _Float16 Qs[128 * 72];  // [m][d] stride 72 halves (144B)
  __shared__ __align__(16) _Float16 Ks[128 * 72];  // [n][d]

  const int tid   = threadIdx.x;
  const int lane  = tid & 31;
  const int waveM = (tid >> 5) & 3;
  const int waveN = tid >> 7;

  const int tM = (blockIdx.x >> 2) * 128;
  const int tN = (blockIdx.x & 3) * 128;
  const int batch = blockIdx.y;              // b*ENT + e
  const int bb = batch / ENT;

  const _Float16* Qg = qh + ((size_t)batch * SEQ + tM) * DHEAD;
  const _Float16* Kg = kh + ((size_t)batch * SEQ + tN) * DHEAD;

#ifdef HAVE_TDM
  if (tid < 32) {                     // wave 0 drives the TDM
    tdm_load_tile_pad(Qg, Qs);
    tdm_load_tile_pad(Kg, Ks);
    __builtin_amdgcn_s_wait_tensorcnt(0);
  }
  __syncthreads();
#else
  // Fallback: stage 128x64 halves each via 16B vector copies
  #pragma unroll
  for (int i = 0; i < 4; ++i) {
    int idx = tid + i * 256;        // 0..1023
    int r   = idx >> 3;
    int c8  = (idx & 7) * 8;        // half offset, 16B aligned
    *(uint4*)(&Qs[r * 72 + c8]) = *(const uint4*)(Qg + (size_t)r * DHEAD + c8);
    *(uint4*)(&Ks[r * 72 + c8]) = *(const uint4*)(Kg + (size_t)r * DHEAD + c8);
  }
  __syncthreads();
#endif

  v8f acc[2][4];
  {
    v8f z = {};
    #pragma unroll
    for (int i = 0; i < 2; ++i)
      #pragma unroll
      for (int j = 0; j < 4; ++j) acc[i][j] = z;
  }

  const int rowL = lane & 15;
  const int kSel = (lane >> 4) * 8;

  #pragma unroll
  for (int kc = 0; kc < DHEAD; kc += 32) {
    v16h a[2], b[4];
    #pragma unroll
    for (int mt = 0; mt < 2; ++mt)
      a[mt] = load_frag(&Qs[(waveM * 32 + mt * 16 + rowL) * 72], kc + kSel);
    #pragma unroll
    for (int nt = 0; nt < 4; ++nt)
      b[nt] = load_frag(&Ks[(waveN * 64 + nt * 16 + rowL) * 72], kc + kSel);
    #pragma unroll
    for (int mt = 0; mt < 2; ++mt)
      #pragma unroll
      for (int nt = 0; nt < 4; ++nt)
        acc[mt][nt] = wmma16(a[mt], b[nt], acc[mt][nt]);
  }

  // Epilogue: pad mask, causal tril mask, scale 1/sqrt(64)
  float* outB = logits + (size_t)batch * SEQ * SEQ;
  #pragma unroll
  for (int mt = 0; mt < 2; ++mt) {
    #pragma unroll
    for (int nt = 0; nt < 4; ++nt) {
      const int ng = tN + waveN * 64 + nt * 16 + rowL;
      const float pad = (float)amask[bb * SEQ + ng];
      const float padSub = (1.0f - pad) * BIGC;
      #pragma unroll
      for (int r = 0; r < 8; ++r) {
        const int mg = tM + waveM * 32 + mt * 16 + ((lane >> 4) << 3) + r;
        float v = acc[mt][nt][r] * pad - padSub;
        if (mg > ng) v -= BIGC;
        outB[(size_t)mg * SEQ + ng] = v * 0.125f;
      }
    }
  }
}

// ---------------------------------------------------------------------------
// Loss: per-row (144 rows of 512*512) online logsumexp, then block merge.
// ---------------------------------------------------------------------------
__device__ __forceinline__ void lse_add(float& m, float& s, float x) {
  if (x > m) { s = s * __expf(m - x) + 1.0f; m = x; }
  else       { s += __expf(x - m); }
}
__device__ __forceinline__ void lse_merge(float& m, float& s, float m2, float s2) {
  if (m2 == -__builtin_inff()) return;
  if (m2 > m) { s = s * __expf(m - m2) + s2; m = m2; }
  else        { s += s2 * __expf(m2 - m); }
}
__device__ __forceinline__ float log1pexp(float x) {   // logaddexp(0, x)
  return (x > 0.0f) ? x + log1pf(__expf(-x)) : log1pf(__expf(x));
}

__global__ __launch_bounds__(256) void row_loss_kernel(
    const float* __restrict__ logits, const int* __restrict__ labels,
    float* __restrict__ rowloss)
{
  __shared__ float nm[256], ns[256], pm[256], ps[256];
  const int row = blockIdx.x;
  const int tid = threadIdx.x;
  const float* t = logits + (size_t)row * (SEQ * SEQ);
  const int*   y = labels + (size_t)row * (SEQ * SEQ);

  float mN = -__builtin_inff(), sN = 0.0f;
  float mP = -__builtin_inff(), sP = 0.0f;
  for (int i = tid; i < SEQ * SEQ; i += 256) {
    const float yf = (float)y[i];
    const float yp = (1.0f - 2.0f * yf) * t[i];
    lse_add(mN, sN, yp - yf * BIGC);
    lse_add(mP, sP, yp - (1.0f - yf) * BIGC);
  }
  nm[tid] = mN; ns[tid] = sN; pm[tid] = mP; ps[tid] = sP;
  __syncthreads();
  for (int off = 128; off > 0; off >>= 1) {
    if (tid < off) {
      lse_merge(nm[tid], ns[tid], nm[tid + off], ns[tid + off]);
      lse_merge(pm[tid], ps[tid], pm[tid + off], ps[tid + off]);
    }
    __syncthreads();
  }
  if (tid == 0) {
    float lseN = nm[0] + __logf(ns[0]);
    float lseP = pm[0] + __logf(ps[0]);
    rowloss[row] = log1pexp(lseN) + log1pexp(lseP);
  }
}

__global__ void final_loss_kernel(const float* __restrict__ rowloss,
                                  float* __restrict__ out)
{
  if (threadIdx.x == 0 && blockIdx.x == 0) {
    float s = 0.0f;
    for (int i = 0; i < NB * ENT; ++i) s += rowloss[i];   // fixed order: deterministic
    out[0] = s * (1.0f / (float)(NB * ENT));
  }
}

// ---------------------------------------------------------------------------
extern "C" void kernel_launch(void* const* d_in, const int* in_sizes, int n_in,
                              void* d_out, int out_size, void* d_ws, size_t ws_size,
                              hipStream_t stream) {
  const float* X      = (const float*)d_in[0];  // (16,512,768)
  const int*   amask  = (const int*)d_in[1];    // (16,512)
  const int*   labels = (const int*)d_in[2];    // (16,9,512,512)
  const float* W      = (const float*)d_in[3];  // (768,1152)
  const float* bias   = (const float*)d_in[4];  // (1152)
  float* out = (float*)d_out;                   // [0]=loss, [1..]=temp_logits

  const size_t QK_ELEMS = (size_t)NB * ENT * SEQ * DHEAD;  // 4,718,592 halves each
  _Float16* qh = (_Float16*)d_ws;
  _Float16* kh = qh + QK_ELEMS;
  float* rowloss = (float*)(kh + QK_ELEMS);

  gemm1_rope_kernel<<<dim3(64, 9), 256, 0, stream>>>(X, W, bias, qh, kh);
  qk_logits_kernel<<<dim3(16, NB * ENT), 256, 0, stream>>>(qh, kh, amask, out + 1);
  row_loss_kernel<<<NB * ENT, 256, 0, stream>>>(out + 1, labels, rowloss);
  final_loss_kernel<<<1, 32, 0, stream>>>(rowloss, out);

  (void)in_sizes; (void)n_in; (void)out_size; (void)ws_size;
}
